// VarianceAdaptor_27023934226644
// MI455X (gfx1250) — matrix operand order repaired and checked by reading, not compiled, not run on hardware
//
#include <hip/hip_runtime.h>
#include <hip/hip_bf16.h>
#include <math.h>

// ---------------------------------------------------------------------------
// VarianceAdaptor for MI455X (gfx1250, wave32, WMMA 16x16x32 f16 -> f32 acc)
// N=32, L=512, T=3072, D=256
// ---------------------------------------------------------------------------

#define NN 32
#define LL 512
#define TT 3072
#define DD 256

typedef __attribute__((ext_vector_type(16))) _Float16 v16h;
typedef __attribute__((ext_vector_type(8)))  float    v8f;
typedef __attribute__((ext_vector_type(8)))  unsigned int v8u;

static __device__ __forceinline__ unsigned short f2h_bits(float f) {
  _Float16 h = (_Float16)f;
  return __builtin_bit_cast(unsigned short, h);
}

// ---------------------------------------------------------------------------
// Pack conv weight [O=256][I=256][K=3] fp32 into WMMA B-fragment layout f16,
// with the 8 fragment dwords CONTIGUOUS per lane so the conv kernel loads
// each lane's fragment as 2x global_load_b128:
//   frag[(((kt*8+kc)*16 + ntile)*32 + lane)*8 + r]
// B order used here (contiguous k, same as A): lane&15 = column N, lane>>4 picks K half:
//   VGPR r holds K = 16*(lane>>4) + 2r, 2r+1
// ---------------------------------------------------------------------------
__global__ void k_pack_w(const float* __restrict__ w, unsigned int* __restrict__ frag) {
  int f = blockIdx.x * 256 + threadIdx.x;          // < 3*8*16*32*8 = 98304
  int r    = f & 7;
  int lane = (f >> 3) & 31;
  int ntl  = (f >> 8) & 15;
  int kc   = (f >> 12) & 7;
  int kt   = f >> 15;
  int o = ntl * 16 + (lane & 15);
  int i = kc * 32 + 2 * r + 16 * (lane >> 4);
  unsigned int lo = f2h_bits(w[(size_t)o * 768 + (size_t)i * 3 + kt]);
  unsigned int hi = f2h_bits(w[(size_t)o * 768 + (size_t)(i + 1) * 3 + kt]);
  frag[f] = lo | (hi << 16);
}

// fp32 -> f16 elementwise
__global__ void k_cast_f16(const float* __restrict__ src, _Float16* __restrict__ dst, int n) {
  int i = blockIdx.x * 256 + threadIdx.x;
  if (i < n) dst[i] = (_Float16)src[i];
}

// per-batch inclusive cumsum of durations [N, L=512]
__global__ void k_cumsum(const int* __restrict__ dur, int* __restrict__ cum) {
  __shared__ int s[LL];
  int n = blockIdx.x, i = threadIdx.x;
  s[i] = dur[n * LL + i];
  __syncthreads();
  for (int off = 1; off < LL; off <<= 1) {
    int v = (i >= off) ? s[i - off] : 0;
    __syncthreads();
    s[i] += v;
    __syncthreads();
  }
  cum[n * LL + i] = s[i];
}

// idx[n,t] = searchsorted_right(cum[n], t) clipped; -1 if t >= total
__global__ void k_build_idx(const int* __restrict__ cum, int* __restrict__ idx) {
  int g = blockIdx.x * 256 + threadIdx.x;          // < N*T
  int n = g / TT, t = g - n * TT;
  const int* c = cum + n * LL;
  int total = c[LL - 1];
  int id = -1;
  if (t < total) {
    int lo = 0, hi = LL;
    while (lo < hi) { int mid = (lo + hi) >> 1; if (c[mid] <= t) lo = mid + 1; else hi = mid; }
    id = lo < (LL - 1) ? lo : (LL - 1);
  }
  idx[g] = id;
}

// xe_h[row] = xh[n, idx[row]] or 0   (dword granularity, 128 dwords/row)
__global__ void k_gather(const _Float16* __restrict__ xh, const int* __restrict__ idx,
                         _Float16* __restrict__ xeh) {
  int row = blockIdx.x;
  int id = idx[row];
  int n = row / TT;
  const unsigned int* src = (const unsigned int*)xh;
  unsigned int* dst = (unsigned int*)xeh;
  unsigned int v = 0u;
  if (id >= 0) v = src[((size_t)n * LL + id) * 128 + threadIdx.x];
  dst[(size_t)row * 128 + threadIdx.x] = v;
}

// ---------------------------------------------------------------------------
// Fused conv1d(k=3, same-pad) + bias + ReLU + LayerNorm, f16 in/out, f32 acc.
// One 512-thread WG (16 waves) -> 32 rows x 256 output channels.
// Wave w owns output columns [16w, 16w+16) for two 16-row M-tiles that share
// each B fragment: 3 taps x 8 K-chunks x 2 M-tiles = 48 WMMAs per wave.
// ---------------------------------------------------------------------------
#define AROW 132   // dwords per staged LDS row (128 + 4 pad to break bank conflicts)
#define TROW 260   // floats per relu-tile row

__global__ __launch_bounds__(512) void k_conv_ln(
    const _Float16* __restrict__ inb,          // [B, R, 256] f16
    const unsigned int* __restrict__ wfrag,    // packed B fragments (98304 dwords)
    const float* __restrict__ bias,
    const float* __restrict__ gamma,
    const float* __restrict__ beta,
    _Float16* __restrict__ outb,               // [B, R, 256] f16
    int R)
{
  __shared__ unsigned int sA[34 * AROW];       // 34 staged rows, f16 pairs (~17.5 KB)
  __shared__ float sT[32 * TROW];              // relu tile fp32 (~32.5 KB)

  int tilesPerBatch = R >> 5;
  int b  = blockIdx.x / tilesPerBatch;
  int t0 = (blockIdx.x % tilesPerBatch) << 5;

  const unsigned int* inRow = (const unsigned int*)inb + (size_t)b * R * 128;

  // stage rows t0-1 .. t0+32 (zero-pad at sequence edges)
  for (int i = threadIdx.x; i < 34 * 128; i += 512) {
    int rl = i >> 7, c = i & 127;
    int gr = t0 - 1 + rl;
    unsigned int v = 0u;
    if (gr >= 0 && gr < R) v = inRow[(size_t)gr * 128 + c];
    sA[rl * AROW + c] = v;
  }
  __syncthreads();

  int lane = threadIdx.x & 31;
  int wv   = threadIdx.x >> 5;                 // 0..15 -> column tile
  int ncol = lane & 15;
  int hi   = lane >> 4;

  v8f acc0, acc1;
#pragma unroll
  for (int r = 0; r < 8; ++r) { acc0[r] = 0.0f; acc1[r] = 0.0f; }

  for (int kt = 0; kt < 3; ++kt) {
    int rowBase0 = (ncol + kt) * AROW;         // M-tile 0: rows t0 + 0..15
    int rowBase1 = (ncol + 16 + kt) * AROW;    // M-tile 1: rows t0 + 16..31
    for (int kc = 0; kc < 8; ++kc) {
      v8u av0, av1;
#pragma unroll
      for (int r = 0; r < 8; ++r) {            // A layout (16-bit 16x32)
        int k0 = 2 * r + 16 * hi;
        int dk = kc * 16 + (k0 >> 1);
        av0[r] = sA[rowBase0 + dk];
        av1[r] = sA[rowBase1 + dk];
      }
      // one 32B contiguous fragment per lane -> 2x global_load_b128
      v8u bv = *(const v8u*)(wfrag + ((((kt * 8 + kc) * 16 + wv) * 32 + lane) << 3));
      v16h bm = __builtin_bit_cast(v16h, bv);
      v16h a0 = __builtin_bit_cast(v16h, av0);
      v16h a1 = __builtin_bit_cast(v16h, av1);
      acc0 = __builtin_amdgcn_wmma_f32_16x16x32_f16(false, a0, false, bm,
                                                    (short)0, acc0, false, false);
      acc1 = __builtin_amdgcn_wmma_f32_16x16x32_f16(false, a1, false, bm,
                                                    (short)0, acc1, false, false);
    }
  }

  // bias + relu into LDS tile (C layout: col = lane&15 + 16*wv, row = r + 8*hi)
  int ocol = wv * 16 + ncol;
  float bvs = bias[ocol];
#pragma unroll
  for (int r = 0; r < 8; ++r) {
    float y0 = acc0[r] + bvs;
    float y1 = acc1[r] + bvs;
    sT[(r + 8 * hi) * TROW + ocol]      = y0 > 0.0f ? y0 : 0.0f;
    sT[(16 + r + 8 * hi) * TROW + ocol] = y1 > 0.0f ? y1 : 0.0f;
  }
  __syncthreads();

  // LayerNorm: wave wv normalizes rows wv and wv+16 (256 channels, 8 per lane)
#pragma unroll
  for (int half = 0; half < 2; ++half) {
    int m = wv + 16 * half;
    float vals[8];
    float s = 0.0f, ss = 0.0f;
#pragma unroll
    for (int j = 0; j < 8; ++j) {
      float v = sT[m * TROW + lane + 32 * j];
      vals[j] = v; s += v; ss += v * v;
    }
    for (int off = 16; off > 0; off >>= 1) {
      s  += __shfl_xor(s, off, 32);
      ss += __shfl_xor(ss, off, 32);
    }
    float mu  = s * (1.0f / 256.0f);
    float var = ss * (1.0f / 256.0f) - mu * mu;
    float inv = rsqrtf(var + 1e-5f);
    size_t orow = ((size_t)b * R + (t0 + m)) * 256;
#pragma unroll
    for (int j = 0; j < 8; ++j) {
      int o = lane + 32 * j;
      float v = (vals[j] - mu) * inv * gamma[o] + beta[o];
      outb[orow + o] = (_Float16)v;
    }
  }
}

// out[row] = fb + dot(h2[row], fw)   (one wave per row)
__global__ void k_rowdot(const _Float16* __restrict__ h2, const float* __restrict__ fw,
                         const float* __restrict__ fb, float* __restrict__ out, int rows) {
  int row  = blockIdx.x * 8 + (threadIdx.x >> 5);
  int lane = threadIdx.x & 31;
  if (row >= rows) return;
  const _Float16* hp = h2 + (size_t)row * 256;
  float s = 0.0f;
#pragma unroll
  for (int j = 0; j < 8; ++j) {
    int o = lane + 32 * j;
    s += (float)hp[o] * fw[o];
  }
  for (int off = 16; off > 0; off >>= 1) s += __shfl_xor(s, off, 32);
  if (lane == 0) out[row] = s + fb[0];
}

// pos[n,t] = t < L ? exp(log_dur[n,t]) : 0
__global__ void k_pos(const float* __restrict__ ldur, float* __restrict__ pos) {
  int g = blockIdx.x * 256 + threadIdx.x;      // < N*T
  int n = g / TT, t = g - n * TT;
  pos[g] = (t < LL) ? expf(ldur[n * LL + t]) : 0.0f;
}

// out[n,t,:] = pitch_emb[pq] + energy_emb[eq] + xe   (xe re-gathered from x)
__global__ void k_combine(const float* __restrict__ x, const int* __restrict__ idx,
                          const float* __restrict__ pitch, const float* __restrict__ energy,
                          const float* __restrict__ pemb, const float* __restrict__ eemb,
                          float* __restrict__ out) {
  __shared__ int sPQ, sEQ, sID;
  int row = blockIdx.x;
  if (threadIdx.x == 0) {
    const float stepP = logf(801.0f) / 255.0f;   // linspace(0, log(801), 256)[:-1]
    const float stepE = 300.0f / 255.0f;         // linspace(0, 300, 256)[:-1]
    float vp = log1pf(pitch[row]);
    int lo = 0, hi = 255;
    while (lo < hi) { int mid = (lo + hi) >> 1; if ((float)mid * stepP < vp) lo = mid + 1; else hi = mid; }
    sPQ = lo;
    float ve = energy[row];
    lo = 0; hi = 255;
    while (lo < hi) { int mid = (lo + hi) >> 1; if ((float)mid * stepE < ve) lo = mid + 1; else hi = mid; }
    sEQ = lo;
    sID = idx[row];
  }
  __syncthreads();
  int n = row / TT;
  int d = threadIdx.x;
  float xv = (sID >= 0) ? x[((size_t)n * LL + sID) * DD + d] : 0.0f;
  out[(size_t)row * DD + d] = pemb[sPQ * DD + d] + eemb[sEQ * DD + d] + xv;
}

// ---------------------------------------------------------------------------
extern "C" void kernel_launch(void* const* d_in, const int* in_sizes, int n_in,
                              void* d_out, int out_size, void* d_ws, size_t ws_size,
                              hipStream_t stream) {
  (void)in_sizes; (void)n_in; (void)out_size; (void)ws_size;

  const float* x   = (const float*)d_in[0];
  const int*   dur = (const int*)d_in[1];
  // predictor blocks: dp=3..12, pp=13..22, ep=23..32 (c1w,c1b,g1,b1,c2w,c2b,g2,b2,fw,fb)
  const float* dp_c1w=(const float*)d_in[3],  *dp_c1b=(const float*)d_in[4];
  const float* dp_g1 =(const float*)d_in[5],  *dp_b1 =(const float*)d_in[6];
  const float* dp_c2w=(const float*)d_in[7],  *dp_c2b=(const float*)d_in[8];
  const float* dp_g2 =(const float*)d_in[9],  *dp_b2 =(const float*)d_in[10];
  const float* dp_fw =(const float*)d_in[11], *dp_fb =(const float*)d_in[12];
  const float* pp_c1w=(const float*)d_in[13], *pp_c1b=(const float*)d_in[14];
  const float* pp_g1 =(const float*)d_in[15], *pp_b1 =(const float*)d_in[16];
  const float* pp_c2w=(const float*)d_in[17], *pp_c2b=(const float*)d_in[18];
  const float* pp_g2 =(const float*)d_in[19], *pp_b2 =(const float*)d_in[20];
  const float* pp_fw =(const float*)d_in[21], *pp_fb =(const float*)d_in[22];
  const float* ep_c1w=(const float*)d_in[23], *ep_c1b=(const float*)d_in[24];
  const float* ep_g1 =(const float*)d_in[25], *ep_b1 =(const float*)d_in[26];
  const float* ep_c2w=(const float*)d_in[27], *ep_c2b=(const float*)d_in[28];
  const float* ep_g2 =(const float*)d_in[29], *ep_b2 =(const float*)d_in[30];
  const float* ep_fw =(const float*)d_in[31], *ep_fb =(const float*)d_in[32];
  const float* pemb  =(const float*)d_in[33];
  const float* eemb  =(const float*)d_in[34];

  // workspace layout
  char* ws = (char*)d_ws;
  const size_t FRAGN = 98304;                         // dwords per packed weight
  unsigned int* wfrag = (unsigned int*)ws;            // 6 * 98304 u32 = 2.25 MB
  _Float16* xh  = (_Float16*)(ws + 2359296);          // x f16   (8 MB)
  int* cum      = (int*)(ws + 10747904);              // 64 KB
  int* idx      = (int*)(ws + 10813440);              // 384 KB
  _Float16* xeh = (_Float16*)(ws + 11206656);         // xe f16  (48 MB)
  _Float16* h1  = (_Float16*)(ws + 61538304);         // 48 MB
  _Float16* h2  = (_Float16*)(ws + 111869952);        // 48 MB

  // output layout (concatenated tuple)
  float* out    = (float*)d_out;
  float* pos    = out + (size_t)NN * TT * DD;         // 25165824
  float* ldur   = pos + (size_t)NN * TT;              // +98304
  float* pitch  = ldur + (size_t)NN * LL;             // +16384
  float* energy = pitch + (size_t)NN * TT;            // +98304

  // 1) pack the 6 conv weights into WMMA B-fragment layout
  k_pack_w<<<384, 256, 0, stream>>>(dp_c1w, wfrag + 0 * FRAGN);
  k_pack_w<<<384, 256, 0, stream>>>(dp_c2w, wfrag + 1 * FRAGN);
  k_pack_w<<<384, 256, 0, stream>>>(pp_c1w, wfrag + 2 * FRAGN);
  k_pack_w<<<384, 256, 0, stream>>>(pp_c2w, wfrag + 3 * FRAGN);
  k_pack_w<<<384, 256, 0, stream>>>(ep_c1w, wfrag + 4 * FRAGN);
  k_pack_w<<<384, 256, 0, stream>>>(ep_c2w, wfrag + 5 * FRAGN);

  // 2) cast x to f16; length-regulate
  k_cast_f16<<<16384, 256, 0, stream>>>(x, xh, NN * LL * DD);
  k_cumsum<<<NN, LL, 0, stream>>>(dur, cum);
  k_build_idx<<<NN * TT / 256, 256, 0, stream>>>(cum, idx);
  k_gather<<<NN * TT, 128, 0, stream>>>(xh, idx, xeh);

  // 3) duration predictor on x  (rows = N*L = 16384, 32 rows/WG)
  k_conv_ln<<<512, 512, 0, stream>>>(xh, wfrag + 0 * FRAGN, dp_c1b, dp_g1, dp_b1, h1, LL);
  k_conv_ln<<<512, 512, 0, stream>>>(h1, wfrag + 1 * FRAGN, dp_c2b, dp_g2, dp_b2, h2, LL);
  k_rowdot<<<2048, 256, 0, stream>>>(h2, dp_fw, dp_fb, ldur, NN * LL);
  k_pos<<<NN * TT / 256, 256, 0, stream>>>(ldur, pos);

  // 4) pitch predictor on xe  (rows = N*T = 98304, 32 rows/WG)
  k_conv_ln<<<3072, 512, 0, stream>>>(xeh, wfrag + 2 * FRAGN, pp_c1b, pp_g1, pp_b1, h1, TT);
  k_conv_ln<<<3072, 512, 0, stream>>>(h1,  wfrag + 3 * FRAGN, pp_c2b, pp_g2, pp_b2, h2, TT);
  k_rowdot<<<12288, 256, 0, stream>>>(h2, pp_fw, pp_fb, pitch, NN * TT);

  // 5) energy predictor on xe
  k_conv_ln<<<3072, 512, 0, stream>>>(xeh, wfrag + 4 * FRAGN, ep_c1b, ep_g1, ep_b1, h1, TT);
  k_conv_ln<<<3072, 512, 0, stream>>>(h1,  wfrag + 5 * FRAGN, ep_c2b, ep_g2, ep_b2, h2, TT);
  k_rowdot<<<12288, 256, 0, stream>>>(h2, ep_fw, ep_fb, energy, NN * TT);

  // 6) bucketize + embedding add + xe
  k_combine<<<NN * TT, 256, 0, stream>>>(x, idx, pitch, energy, pemb, eemb, out);
}
